// EGNNModelTorch_31653908971780
// MI455X (gfx1250) — matrix operand order repaired and checked by reading, not compiled, not run on hardware
//
#include <hip/hip_runtime.h>

typedef __attribute__((ext_vector_type(16))) _Float16 v16h;
typedef __attribute__((ext_vector_type(8)))  float    v8f;

#define D_IN   16
#define D_H    128
#define D_MSG  64
#define D_E    16
#define N_LAYERS 6

__device__ __forceinline__ float silu_f(float v) { return v / (1.0f + __expf(-v)); }

__device__ __forceinline__ v8f zero8() {
  v8f z = {0.f, 0.f, 0.f, 0.f, 0.f, 0.f, 0.f, 0.f};
  return z;
}

__device__ __forceinline__ v8f wmma_f16(v16h a, v16h b, v8f c) {
  // D = A(16x32 f16) * B(32x16 f16) + C(16x16 f32)
  return __builtin_amdgcn_wmma_f32_16x16x32_f16(false, a, false, b, (short)0, c, false, false);
}

// A-matrix pack (16x32 f16 tile, K slice [kbase,kbase+32)).
// Lane l holds row (l&15). halves 0..7 -> K = kbase+hi8+{0..7},
// halves 8..15 -> K = kbase+16+hi8+{0..7}  (hi8 = (l>=16)?8:0).
__device__ __forceinline__ v16h packA_f16(const _Float16* row, int kbase, int hi8) {
  v16h a;
#pragma unroll
  for (int q = 0; q < 8; ++q) {
    a[q]     = row[kbase + hi8 + q];
    a[q + 8] = row[kbase + 16 + hi8 + q];
  }
  return a;
}

__device__ __forceinline__ v16h packA_f32(const float* row, int kbase, int hi8) {
  v16h a;
#pragma unroll
  for (int q = 0; q < 8; ++q) {
    a[q]     = (_Float16)row[kbase + hi8 + q];
    a[q + 8] = (_Float16)row[kbase + 16 + hi8 + q];
  }
  return a;
}

// Load a pre-packed B fragment: one contiguous 32-byte chunk per lane.
// tile = kt * (N/16) + nb
__device__ __forceinline__ v16h loadB(const _Float16* __restrict__ P, int tile, int lane) {
  return *(const v16h*)(P + ((size_t)tile * 32 + lane) * 16);
}

// Store accumulator blocks (C layout: lane -> (m = r + 8*hi, n = nb*16 + (l&15)))
// into a row-major LDS tile, adding bias and optionally applying SiLU.
template <int NB>
__device__ __forceinline__ void store_acc(float* buf, int ldn, const v8f* acc,
                                          const float* bias, int lane, bool dosilu) {
  const int n0 = lane & 15, hi = lane >> 4;
#pragma unroll
  for (int nb = 0; nb < NB; ++nb) {
    const int n = nb * 16 + n0;
    const float bb = bias[n];
#pragma unroll
    for (int r = 0; r < 8; ++r) {
      float v = acc[nb][r] + bb;
      buf[(r + hi * 8) * ldn + n] = dosilu ? silu_f(v) : v;
    }
  }
}

__device__ __forceinline__ float tail_val(int kk, float sq, const float* efr) {
  if (kk == 0) return sq;
  if (kk <= D_E) return efr[kk - 1];
  return 0.0f;
}

// ---------------------------------------------------------------------------
// Pre-pack a row-major f32 weight [K x N] into wave32 WMMA B-fragment order:
//   P[((kt*(N/16)+nb)*32 + lane)*16 + q] = f16( W[(kt*32+(q&8)*2+(lane>>4)*8+(q&7))*N + nb*16+(lane&15)] )
// Rows k >= K are zero (K-tail padding).
// ---------------------------------------------------------------------------
__global__ void egnn_pack_b_kernel(const float* __restrict__ W, _Float16* __restrict__ P,
                                   int K, int N, int total) {
  const int p = blockIdx.x * blockDim.x + threadIdx.x;
  if (p >= total) return;
  const int q = p & 15;
  const int lane = (p >> 4) & 31;
  const int tile = p >> 9;
  const int NB = N >> 4;
  const int nb = tile % NB;
  const int kt = tile / NB;
  const int k = kt * 32 + (q & 8) * 2 + (lane >> 4) * 8 + (q & 7);
  const int col = nb * 16 + (lane & 15);
  P[p] = (k < K) ? (_Float16)W[(size_t)k * N + col] : (_Float16)0.0f;
}

// ---------------------------------------------------------------------------
// Encoder: h = silu(nf @ w1 + b1) @ w2 + b2     (one node per 128-thread block)
// ---------------------------------------------------------------------------
__global__ __launch_bounds__(128) void egnn_encoder_kernel(
    const float* __restrict__ nf,
    const float* __restrict__ w1, const float* __restrict__ b1,
    const float* __restrict__ w2, const float* __restrict__ b2,
    float* __restrict__ h, _Float16* __restrict__ hf) {
  __shared__ float s_in[D_IN];
  __shared__ float s_h1[D_H];
  const int node = blockIdx.x;
  const int j = threadIdx.x;
  if (j < D_IN) s_in[j] = nf[node * D_IN + j];
  __syncthreads();
  float v = b1[j];
#pragma unroll
  for (int k = 0; k < D_IN; ++k) v += s_in[k] * w1[k * D_H + j];
  s_h1[j] = silu_f(v);
  __syncthreads();
  float o = b2[j];
  for (int k = 0; k < D_H; ++k) o += s_h1[k] * w2[k * D_H + j];
  const size_t idx = (size_t)node * D_H + j;
  h[idx] = o;
  hf[idx] = (_Float16)o;
}

// ---------------------------------------------------------------------------
// Edge kernel: one wave = 16 edges. Edge MLP + coord-weight MLP via WMMA,
// then fp32 atomic scatter of messages / coord updates / counts.
// B operands come from pre-packed f16 fragment arrays.
// ---------------------------------------------------------------------------
__global__ __launch_bounds__(32) void egnn_edge_kernel(
    const int* __restrict__ ei, const float* __restrict__ ef,
    const _Float16* __restrict__ hf, const float* __restrict__ x,
    const _Float16* __restrict__ pew1, const float* __restrict__ eb1,
    const _Float16* __restrict__ pew2, const float* __restrict__ eb2,
    const _Float16* __restrict__ pxw1, const float* __restrict__ xb1,
    const float* __restrict__ xw2, const float* __restrict__ xb2,
    float* __restrict__ agg, float* __restrict__ cnt, float* __restrict__ cup,
    int E) {
  __shared__ float s_buf[16 * D_MSG];   // m1, then msg
  __shared__ float s_m2[16 * D_MSG];
  __shared__ float s_diff[16 * 3];
  __shared__ float s_sq[16];
  __shared__ int s_src[16], s_dst[16], s_ok[16];

  const int lane = threadIdx.x;
  const int m = lane & 15;
  const int hi = lane >> 4;
  const int hi8 = hi * 8;
  const int e0 = blockIdx.x * 16;

  if (hi == 0) {
    const int e = e0 + m;
    const int ok = (e < E) ? 1 : 0;
    const int ec = ok ? e : (E - 1);
    const int s = ei[ec];
    const int d = ei[E + ec];
    s_src[m] = s; s_dst[m] = d; s_ok[m] = ok;
    const float dx = x[s * 3 + 0] - x[d * 3 + 0];
    const float dy = x[s * 3 + 1] - x[d * 3 + 1];
    const float dz = x[s * 3 + 2] - x[d * 3 + 2];
    s_diff[m * 3 + 0] = dx; s_diff[m * 3 + 1] = dy; s_diff[m * 3 + 2] = dz;
    s_sq[m] = dx * dx + dy * dy + dz * dz;
  }
  __syncthreads();

  const int ecl = (e0 + m < E) ? (e0 + m) : (E - 1);
  const float* efr = ef + (size_t)ecl * D_E;
  const float sq = s_sq[m];
  const _Float16* rs = hf + (size_t)s_src[m] * D_H;
  const _Float16* rd = hf + (size_t)s_dst[m] * D_H;

  v8f acc[4];
#pragma unroll
  for (int nb = 0; nb < 4; ++nb) acc[nb] = zero8();

  // K = 0..127 : h[src] @ ew1 rows 0..127  (K-tiles 0..3)
#pragma unroll
  for (int ks = 0; ks < 4; ++ks) {
    v16h a = packA_f16(rs, ks * 32, hi8);
#pragma unroll
    for (int nb = 0; nb < 4; ++nb)
      acc[nb] = wmma_f16(a, loadB(pew1, ks * 4 + nb, lane), acc[nb]);
  }
  // K = 128..255 : h[dst] @ ew1 rows 128..255  (K-tiles 4..7)
#pragma unroll
  for (int ks = 0; ks < 4; ++ks) {
    v16h a = packA_f16(rd, ks * 32, hi8);
#pragma unroll
    for (int nb = 0; nb < 4; ++nb)
      acc[nb] = wmma_f16(a, loadB(pew1, (4 + ks) * 4 + nb, lane), acc[nb]);
  }
  // Tail: rows 256..272 = [sq, edge_feats], zero-padded (K-tile 8)
  {
    v16h a;
#pragma unroll
    for (int q = 0; q < 8; ++q) {
      a[q]     = (_Float16)tail_val(hi8 + q, sq, efr);
      a[q + 8] = (_Float16)tail_val(16 + hi8 + q, sq, efr);
    }
#pragma unroll
    for (int nb = 0; nb < 4; ++nb)
      acc[nb] = wmma_f16(a, loadB(pew1, 8 * 4 + nb, lane), acc[nb]);
  }
  store_acc<4>(s_buf, D_MSG, acc, eb1, lane, true);   // m1 = silu(mi@w1+b1)
  __syncthreads();

  // msg = silu(m1 @ ew2 + eb2)
#pragma unroll
  for (int nb = 0; nb < 4; ++nb) acc[nb] = zero8();
#pragma unroll
  for (int ks = 0; ks < 2; ++ks) {
    v16h a = packA_f32(&s_buf[m * D_MSG], ks * 32, hi8);
#pragma unroll
    for (int nb = 0; nb < 4; ++nb)
      acc[nb] = wmma_f16(a, loadB(pew2, ks * 4 + nb, lane), acc[nb]);
  }
  __syncthreads();
  store_acc<4>(s_buf, D_MSG, acc, eb2, lane, true);   // overwrite with msg
  __syncthreads();

  // m2 = silu(msg @ xw1 + xb1)
#pragma unroll
  for (int nb = 0; nb < 4; ++nb) acc[nb] = zero8();
#pragma unroll
  for (int ks = 0; ks < 2; ++ks) {
    v16h a = packA_f32(&s_buf[m * D_MSG], ks * 32, hi8);
#pragma unroll
    for (int nb = 0; nb < 4; ++nb)
      acc[nb] = wmma_f16(a, loadB(pxw1, ks * 4 + nb, lane), acc[nb]);
  }
  store_acc<4>(s_m2, D_MSG, acc, xb1, lane, true);
  __syncthreads();

  // w = m2 @ xw2 + xb2 ; scatter coord update + counts
  if (hi == 0 && s_ok[m]) {
    float wv = xb2[0];
    for (int k = 0; k < D_MSG; ++k) wv += s_m2[m * D_MSG + k] * xw2[k];
    const int s = s_src[m], d = s_dst[m];
    atomicAdd(&cup[s * 3 + 0], s_diff[m * 3 + 0] * wv);
    atomicAdd(&cup[s * 3 + 1], s_diff[m * 3 + 1] * wv);
    atomicAdd(&cup[s * 3 + 2], s_diff[m * 3 + 2] * wv);
    atomicAdd(&cnt[d], 1.0f);
  }
  // Scatter msg into agg[dst]
  for (int i = lane; i < 16 * D_MSG; i += 32) {
    const int mm = i >> 6;
    if (s_ok[mm]) atomicAdd(&agg[(size_t)s_dst[mm] * D_MSG + (i & 63)], s_buf[i]);
  }
}

// ---------------------------------------------------------------------------
// Node kernel: one wave = 16 nodes. dh MLP via WMMA, residual + layer norm.
// ---------------------------------------------------------------------------
__global__ __launch_bounds__(32) void egnn_node_kernel(
    const float* __restrict__ h, const _Float16* __restrict__ hf,
    const float* __restrict__ agg, const float* __restrict__ cnt,
    const _Float16* __restrict__ phw1, const float* __restrict__ hb1,
    const _Float16* __restrict__ phw2, const float* __restrict__ hb2,
    const float* __restrict__ lng, const float* __restrict__ lnb,
    float* __restrict__ hout, _Float16* __restrict__ hfout, int N) {
  __shared__ float s_buf[16 * D_H];
  const int lane = threadIdx.x;
  const int m = lane & 15, hi = lane >> 4, hi8 = hi * 8, n0 = m;
  const int nodebase = blockIdx.x * 16;
  const int node = (nodebase + m < N) ? (nodebase + m) : (N - 1);
  const _Float16* hr = hf + (size_t)node * D_H;
  const float invc = 1.0f / fmaxf(cnt[node], 1.0f);
  const float* ar = agg + (size_t)node * D_MSG;

  v8f acc[8];
#pragma unroll
  for (int nb = 0; nb < 8; ++nb) acc[nb] = zero8();

  // K = 0..127 : h part of concat(h, agg)   (K-tiles 0..3)
#pragma unroll
  for (int ks = 0; ks < 4; ++ks) {
    v16h a = packA_f16(hr, ks * 32, hi8);
#pragma unroll
    for (int nb = 0; nb < 8; ++nb)
      acc[nb] = wmma_f16(a, loadB(phw1, ks * 8 + nb, lane), acc[nb]);
  }
  // K = 128..191 : agg/count part  (K-tiles 4..5)
#pragma unroll
  for (int ks = 0; ks < 2; ++ks) {
    v16h a;
#pragma unroll
    for (int q = 0; q < 8; ++q) {
      a[q]     = (_Float16)(ar[ks * 32 + hi8 + q] * invc);
      a[q + 8] = (_Float16)(ar[ks * 32 + 16 + hi8 + q] * invc);
    }
#pragma unroll
    for (int nb = 0; nb < 8; ++nb)
      acc[nb] = wmma_f16(a, loadB(phw1, (4 + ks) * 8 + nb, lane), acc[nb]);
  }
  store_acc<8>(s_buf, D_H, acc, hb1, lane, true);   // silu(first layer)
  __syncthreads();

  // dh = m1 @ hw2 + hb2
#pragma unroll
  for (int nb = 0; nb < 8; ++nb) acc[nb] = zero8();
#pragma unroll
  for (int ks = 0; ks < 4; ++ks) {
    v16h a = packA_f32(&s_buf[m * D_H], ks * 32, hi8);
#pragma unroll
    for (int nb = 0; nb < 8; ++nb)
      acc[nb] = wmma_f16(a, loadB(phw2, ks * 8 + nb, lane), acc[nb]);
  }
  __syncthreads();
  // y = h + dh + hb2  -> s_buf
#pragma unroll
  for (int nb = 0; nb < 8; ++nb) {
    const int n = nb * 16 + n0;
    const float bb = hb2[n];
#pragma unroll
    for (int r = 0; r < 8; ++r) {
      const int row = r + hi * 8;
      const int rn = (nodebase + row < N) ? (nodebase + row) : (N - 1);
      s_buf[row * D_H + n] = acc[nb][r] + bb + h[(size_t)rn * D_H + n];
    }
  }
  __syncthreads();
  // Layer norm: lane m normalizes row m
  if (hi == 0 && (nodebase + m) < N) {
    float mu = 0.f;
    for (int k = 0; k < D_H; ++k) mu += s_buf[m * D_H + k];
    mu *= (1.0f / D_H);
    float var = 0.f;
    for (int k = 0; k < D_H; ++k) {
      const float dv = s_buf[m * D_H + k] - mu;
      var += dv * dv;
    }
    var *= (1.0f / D_H);
    const float rstd = rsqrtf(var + 1e-5f);
    const size_t o = (size_t)(nodebase + m) * D_H;
    for (int k = 0; k < D_H; ++k) {
      const float v = (s_buf[m * D_H + k] - mu) * rstd * lng[k] + lnb[k];
      hout[o + k] = v;
      hfout[o + k] = (_Float16)v;
    }
  }
}

// ---------------------------------------------------------------------------
// x += coord_update / count
// ---------------------------------------------------------------------------
__global__ void egnn_coord_kernel(float* __restrict__ x, const float* __restrict__ cup,
                                  const float* __restrict__ cnt, int N) {
  const int i = blockIdx.x * blockDim.x + threadIdx.x;
  if (i < N * 3) {
    const int node = i / 3;
    x[i] += cup[i] / fmaxf(cnt[node], 1.0f);
  }
}

__global__ void egnn_zero_kernel(float* __restrict__ p, long long n) {
  const long long i = (long long)blockIdx.x * blockDim.x + threadIdx.x;
  if (i < n) p[i] = 0.0f;
}

// ---------------------------------------------------------------------------
// Graph mean pooling (atomic) and readout MLP
// ---------------------------------------------------------------------------
__global__ void egnn_pool_kernel(const float* __restrict__ h, const int* __restrict__ batch,
                                 float* __restrict__ gh, float* __restrict__ gc, int N) {
  const int i = blockIdx.x * blockDim.x + threadIdx.x;
  if (i < N * D_H) {
    const int node = i >> 7;
    const int f = i & (D_H - 1);
    const int b = batch[node];
    atomicAdd(&gh[b * D_H + f], h[i]);
    if (f == 0) atomicAdd(&gc[b], 1.0f);
  }
}

__global__ __launch_bounds__(128) void egnn_readout_kernel(
    const float* __restrict__ gh, const float* __restrict__ gc,
    const float* __restrict__ w1, const float* __restrict__ b1,
    const float* __restrict__ w2, const float* __restrict__ b2,
    const float* __restrict__ w3, const float* __restrict__ b3,
    float* __restrict__ out) {
  __shared__ float s_g[D_H], s_r1[D_H], s_r2[D_H / 2];
  const int gidx = blockIdx.x, j = threadIdx.x;
  const float invc = 1.0f / fmaxf(gc[gidx], 1.0f);
  s_g[j] = gh[gidx * D_H + j] * invc;
  __syncthreads();
  float v = b1[j];
  for (int k = 0; k < D_H; ++k) v += s_g[k] * w1[k * D_H + j];
  s_r1[j] = silu_f(v);
  __syncthreads();
  if (j < D_H / 2) {
    float u = b2[j];
    for (int k = 0; k < D_H; ++k) u += s_r1[k] * w2[k * (D_H / 2) + j];
    s_r2[j] = silu_f(u);
  }
  __syncthreads();
  if (j < 2) {
    float o = b3[j];
    for (int k = 0; k < D_H / 2; ++k) o += s_r2[k] * w3[k * 2 + j];
    out[gidx * 2 + j] = o;
  }
}

// ---------------------------------------------------------------------------
extern "C" void kernel_launch(void* const* d_in, const int* in_sizes, int n_in,
                              void* d_out, int out_size, void* d_ws, size_t ws_size,
                              hipStream_t stream) {
  const float* node_feats = (const float*)d_in[0];
  const float* coords     = (const float*)d_in[1];
  const int*   edge_index = (const int*)d_in[2];
  const float* edge_feats = (const float*)d_in[3];
  const int*   batch_idx  = (const int*)d_in[4];
  const float* enc_w1 = (const float*)d_in[5];
  const float* enc_b1 = (const float*)d_in[6];
  const float* enc_w2 = (const float*)d_in[7];
  const float* enc_b2 = (const float*)d_in[8];
  const float* pe_w1 = (const float*)d_in[9];
  const float* pe_b1 = (const float*)d_in[10];
  const float* pe_w2 = (const float*)d_in[11];
  const float* pe_b2 = (const float*)d_in[12];
  const float* ph_w1 = (const float*)d_in[13];
  const float* ph_b1 = (const float*)d_in[14];
  const float* ph_w2 = (const float*)d_in[15];
  const float* ph_b2 = (const float*)d_in[16];
  const float* px_w1 = (const float*)d_in[17];
  const float* px_b1 = (const float*)d_in[18];
  const float* px_w2 = (const float*)d_in[19];
  const float* px_b2 = (const float*)d_in[20];
  const float* ln_g  = (const float*)d_in[21];
  const float* ln_b  = (const float*)d_in[22];
  const float* ro_w1 = (const float*)d_in[23];
  const float* ro_b1 = (const float*)d_in[24];
  const float* ro_w2 = (const float*)d_in[25];
  const float* ro_b2 = (const float*)d_in[26];
  const float* ro_w3 = (const float*)d_in[27];
  const float* ro_b3 = (const float*)d_in[28];

  const int N = in_sizes[0] / D_IN;
  const int E = in_sizes[2] / 2;
  const int G = out_size / 2;

  // Packed-fragment sizes per layer (in halves): Ktiles * (N/16) * 512
  const int SZ_EW1 = 9 * 4 * 512;   // K=273 (9 tiles), N=64
  const int SZ_EW2 = 2 * 4 * 512;   // K=64,  N=64
  const int SZ_XW1 = 2 * 4 * 512;   // K=64,  N=64
  const int SZ_HW1 = 6 * 8 * 512;   // K=192, N=128
  const int SZ_HW2 = 4 * 8 * 512;   // K=128, N=128

  // Workspace layout (256B-aligned bump allocator)
  char* wsp = (char*)d_ws;
  auto alloc = [&](size_t bytes) -> void* {
    void* p = (void*)wsp;
    wsp += (bytes + 255) & ~(size_t)255;
    return p;
  };
  float*    hA  = (float*)alloc((size_t)N * D_H * sizeof(float));
  float*    hB  = (float*)alloc((size_t)N * D_H * sizeof(float));
  _Float16* hfA = (_Float16*)alloc((size_t)N * D_H * sizeof(_Float16));
  _Float16* hfB = (_Float16*)alloc((size_t)N * D_H * sizeof(_Float16));
  float*    agg = (float*)alloc((size_t)N * D_MSG * sizeof(float));
  float*    cnt = (float*)alloc((size_t)N * sizeof(float));
  float*    cup = (float*)alloc((size_t)N * 3 * sizeof(float));
  float*    x   = (float*)alloc((size_t)N * 3 * sizeof(float));
  float*    gh  = (float*)alloc((size_t)G * D_H * sizeof(float));
  float*    gc  = (float*)alloc((size_t)G * sizeof(float));
  _Float16* pew1 = (_Float16*)alloc((size_t)N_LAYERS * SZ_EW1 * sizeof(_Float16));
  _Float16* pew2 = (_Float16*)alloc((size_t)N_LAYERS * SZ_EW2 * sizeof(_Float16));
  _Float16* pxw1 = (_Float16*)alloc((size_t)N_LAYERS * SZ_XW1 * sizeof(_Float16));
  _Float16* phw1 = (_Float16*)alloc((size_t)N_LAYERS * SZ_HW1 * sizeof(_Float16));
  _Float16* phw2 = (_Float16*)alloc((size_t)N_LAYERS * SZ_HW2 * sizeof(_Float16));
  (void)ws_size; (void)n_in;

  // Span covering agg..cup (including alignment gaps) for one-shot zeroing
  const long long zspan1 = (long long)(((char*)cup + (size_t)N * 3 * sizeof(float)) - (char*)agg) / 4;
  const long long zspan2 = (long long)(((char*)gc + (size_t)G * sizeof(float)) - (char*)gh) / 4;

  hipMemcpyAsync(x, coords, (size_t)N * 3 * sizeof(float), hipMemcpyDeviceToDevice, stream);

  // Pre-pack all WMMA B fragments (f16) for all layers.
  for (int l = 0; l < N_LAYERS; ++l) {
    egnn_pack_b_kernel<<<(SZ_EW1 + 255) / 256, 256, 0, stream>>>(
        pe_w1 + (size_t)l * 273 * D_MSG, pew1 + (size_t)l * SZ_EW1, 273, D_MSG, SZ_EW1);
    egnn_pack_b_kernel<<<(SZ_EW2 + 255) / 256, 256, 0, stream>>>(
        pe_w2 + (size_t)l * D_MSG * D_MSG, pew2 + (size_t)l * SZ_EW2, D_MSG, D_MSG, SZ_EW2);
    egnn_pack_b_kernel<<<(SZ_XW1 + 255) / 256, 256, 0, stream>>>(
        px_w1 + (size_t)l * D_MSG * D_MSG, pxw1 + (size_t)l * SZ_XW1, D_MSG, D_MSG, SZ_XW1);
    egnn_pack_b_kernel<<<(SZ_HW1 + 255) / 256, 256, 0, stream>>>(
        ph_w1 + (size_t)l * (D_H + D_MSG) * D_H, phw1 + (size_t)l * SZ_HW1, D_H + D_MSG, D_H, SZ_HW1);
    egnn_pack_b_kernel<<<(SZ_HW2 + 255) / 256, 256, 0, stream>>>(
        ph_w2 + (size_t)l * D_H * D_H, phw2 + (size_t)l * SZ_HW2, D_H, D_H, SZ_HW2);
  }

  egnn_encoder_kernel<<<N, 128, 0, stream>>>(node_feats, enc_w1, enc_b1, enc_w2, enc_b2, hA, hfA);

  float*    hcur = hA;  _Float16* hfcur = hfA;
  float*    hnxt = hB;  _Float16* hfnxt = hfB;

  for (int l = 0; l < N_LAYERS; ++l) {
    egnn_zero_kernel<<<(int)((zspan1 + 255) / 256), 256, 0, stream>>>(agg, zspan1);
    egnn_edge_kernel<<<(E + 15) / 16, 32, 0, stream>>>(
        edge_index, edge_feats, hfcur, x,
        pew1 + (size_t)l * SZ_EW1, pe_b1 + (size_t)l * D_MSG,
        pew2 + (size_t)l * SZ_EW2, pe_b2 + (size_t)l * D_MSG,
        pxw1 + (size_t)l * SZ_XW1, px_b1 + (size_t)l * D_MSG,
        px_w2 + (size_t)l * D_MSG, px_b2 + (size_t)l,
        agg, cnt, cup, E);
    egnn_node_kernel<<<(N + 15) / 16, 32, 0, stream>>>(
        hcur, hfcur, agg, cnt,
        phw1 + (size_t)l * SZ_HW1, ph_b1 + (size_t)l * D_H,
        phw2 + (size_t)l * SZ_HW2, ph_b2 + (size_t)l * D_H,
        ln_g + (size_t)l * D_H, ln_b + (size_t)l * D_H,
        hnxt, hfnxt, N);
    egnn_coord_kernel<<<(N * 3 + 255) / 256, 256, 0, stream>>>(x, cup, cnt, N);
    float* tf = hcur; hcur = hnxt; hnxt = tf;
    _Float16* th = hfcur; hfcur = hfnxt; hfnxt = th;
  }

  egnn_zero_kernel<<<(int)((zspan2 + 255) / 256), 256, 0, stream>>>(gh, zspan2);
  egnn_pool_kernel<<<(N * D_H + 255) / 256, 256, 0, stream>>>(hcur, batch_idx, gh, gc, N);
  egnn_readout_kernel<<<G, 128, 0, stream>>>(gh, gc, ro_w1, ro_b1, ro_w2, ro_b2, ro_w3, ro_b3,
                                             (float*)d_out);
}
